// SoftPatternClassifier_69423851373334
// MI455X (gfx1250) — matrix operand order repaired and checked by reading, not compiled, not run on hardware
//
#include <hip/hip_runtime.h>

// ---------------------------------------------------------------------------
// SoPa classifier for MI455X (gfx1250), round 3.
//   K1: f32 -> bf16 convert + row-pad (300->320) of embeddings
//   K2: f32 -> bf16 convert + row-pad of diags
//   K3: fused gather + bf16 WMMA GEMM + max-plus scan, fully pipelined:
//         async_gather A(i+1)  ||  WMMA GEMM(i) on waves 0-6  ||  scan(i-1) on wave 7
//       using global_load_async_to_lds_b128 (ASYNCcnt) + double-buffered LDS.
//   K4: tiny MLP + log_softmax
// ---------------------------------------------------------------------------

typedef __attribute__((ext_vector_type(16))) __bf16        v16bf;
typedef __attribute__((ext_vector_type(8)))  float         v8f;
typedef __attribute__((ext_vector_type(4)))  unsigned int  v4u;

union FragBF { v4u q[2]; v16bf v; };

#define NEGV (-100.0f)

constexpr int V_ = 50000, E_ = 300, EPAD = 320;
constexpr int Bb = 64, Ll = 512, Pp = 200, Mm = 7;
constexpr int Dd = 2800, Hh = 100;
constexpr int PG  = 8;              // patterns per workgroup
constexpr int DC  = PG * 2 * Mm;    // 112 columns = 7 N-tiles of 16
constexpr int NPG = Pp / PG;        // 25
constexpr int LT  = 16;             // timesteps per tile (WMMA M)
constexpr int NLT = Ll / LT;        // 32

constexpr size_t WS_EMB   = 0;
constexpr size_t WS_DIAG  = (size_t)V_ * EPAD * 2;               // 32,000,000
constexpr size_t WS_SCORE = WS_DIAG + (size_t)Dd * EPAD * 2;     // 33,792,000

// LDS layout (bytes)
constexpr int LDS_A0   = 0;          // [16][320] bf16  (x2, double buffered)
constexpr int LDS_ASTRIDE = 10240;
constexpr int LDS_B    = 20480;      // [112][320] bf16
constexpr int LDS_TS0  = 92160;      // [16][112] f32   (x2, double buffered)
constexpr int LDS_TSSTRIDE = 7168;
constexpr int LDS_BIAS = 106496;     // [112] f32
constexpr int LDS_TOTAL= 106944;

__device__ __forceinline__ unsigned short f2bf(float f) {
  unsigned int u = __float_as_uint(f);
  u += 0x7fffu + ((u >> 16) & 1u);  // round-to-nearest-even
  return (unsigned short)(u >> 16);
}

// gfx1250 async copy global -> LDS (tracked by ASYNCcnt).
// Generic pointers to LDS carry the LDS byte offset in their low 32 bits.
__device__ __forceinline__ void async_b128(void* lds, const void* g) {
  unsigned l = (unsigned)(unsigned long long)lds;
  asm volatile("global_load_async_to_lds_b128 %0, %1, off"
               :: "v"(l), "v"(g) : "memory");
}
__device__ __forceinline__ void async_wait0() {
  asm volatile("s_wait_asynccnt 0x0" ::: "memory");
}

// ---- K1: embeddings f32 [V,300] -> bf16 [V,320] zero-padded ----------------
__global__ void cvt_emb_kernel(const float* __restrict__ in,
                               unsigned short* __restrict__ out) {
  int i = blockIdx.x * blockDim.x + threadIdx.x;
  if (i < V_ * EPAD) {
    int v = i / EPAD, e = i - v * EPAD;
    out[i] = (e < E_) ? f2bf(in[v * E_ + e]) : (unsigned short)0;
  }
}

// ---- K2: diags f32 [D,300] -> bf16 [D,320] zero-padded ---------------------
__global__ void cvt_diag_kernel(const float* __restrict__ in,
                                unsigned short* __restrict__ out) {
  int i = blockIdx.x * blockDim.x + threadIdx.x;
  if (i < Dd * EPAD) {
    int d = i / EPAD, e = i - d * EPAD;
    out[i] = (e < E_) ? f2bf(in[d * E_ + e]) : (unsigned short)0;
  }
}

// ---- K3: fused pipelined gather-GEMM-scan ----------------------------------
__global__ void sopa_kernel(const int* __restrict__ docs,
                            const unsigned short* __restrict__ embB,
                            const unsigned short* __restrict__ diagB,
                            const float* __restrict__ bias,
                            const float* __restrict__ eps,
                            float* __restrict__ scores) {
  const int b  = blockIdx.x;
  const int pg = blockIdx.y;
  const int d0 = pg * DC;

  extern __shared__ char smem[];
  // Runtime-computed LDS pointers (no constant pointer arrays: ld.lld cannot
  // statically relocate addrspacecast-of-LDS initializers).
  auto Ab  = [&](int i) -> unsigned short* {
    return (unsigned short*)(smem + LDS_A0 + i * LDS_ASTRIDE);
  };
  auto TSb = [&](int i) -> float* {
    return (float*)(smem + LDS_TS0 + i * LDS_TSSTRIDE);
  };
  unsigned short* Bs    = (unsigned short*)(smem + LDS_B);
  float*          biasS = (float*)(smem + LDS_BIAS);

  const int tid = threadIdx.x;
  const int wv  = tid >> 5;
  const int ln  = tid & 31;
  const int lw  = ln & 15;
  const int hi  = ln >> 4;
  const bool scanLane = (wv == 7) && (ln < PG);

  // async gather of one 16-row A tile (40 aligned B128 chunks per row)
  auto gather_tile = [&](int lt, unsigned short* Ad) {
    const int r   = tid >> 4;
    const int sub = tid & 15;
    const int tok = docs[b * Ll + lt * LT + r];
    const char* src = (const char*)(embB + (size_t)tok * EPAD);
    char*       dst = (char*)(Ad + r * EPAD);
#pragma unroll
    for (int j3 = 0; j3 < 3; ++j3) {
      const int j = sub + 16 * j3;
      if (j < EPAD * 2 / 16) async_b128(dst + 16 * j, src + 16 * j);
    }
  };

  // ---- preamble: async-preload diag chunk + A tile 0 + bias ----
  {
    const char* src = (const char*)(diagB + (size_t)d0 * EPAD);
    char*       dst = (char*)Bs;
    for (int c = tid; c < DC * EPAD * 2 / 16; c += 256)
      async_b128(dst + 16 * c, src + 16 * c);
  }
  gather_tile(0, Ab(0));
  for (int i = tid; i < DC; i += 256) biasS[i] = bias[d0 + i];

  // ---- scan state: one pattern per lane of wave 7 ----
  float h[Mm], ev[Mm - 1], sacc = NEGV;
  if (scanLane) {
    const int p = pg * PG + ln;
#pragma unroll
    for (int j = 0; j < Mm - 1; ++j) ev[j] = eps[p * (Mm - 1) + j];
    h[0] = 0.0f;
#pragma unroll
    for (int m = 1; m < Mm; ++m) h[m] = NEGV;
  }

  auto scan_tile = [&](const float* TSs) {
#pragma unroll 4
    for (int st = 0; st < LT; ++st) {
      const float* x = &TSs[st * DC + ln * (2 * Mm)];
      float ae[Mm], hn[Mm];
      ae[0] = fmaxf(h[0], NEGV);
#pragma unroll
      for (int m = 1; m < Mm; ++m) ae[m] = fmaxf(h[m], h[m - 1] + ev[m - 1]);
      hn[0] = fmaxf(0.0f, ae[0] + x[0]);
#pragma unroll
      for (int m = 1; m < Mm; ++m)
        hn[m] = fmaxf(ae[m - 1] + x[Mm + m - 1], ae[m] + x[m]);
#pragma unroll
      for (int m = 0; m < Mm; ++m) h[m] = hn[m];
      sacc = fmaxf(sacc, h[Mm - 1]);
    }
  };

  // load fragments for one K step (A and B interleaved halves)
  auto load_frags = [&](const unsigned short* Ad, int n, int e0,
                        FragBF& fa, FragBF& fb) {
    const v4u* pa = (const v4u*)(Ad + lw * EPAD + e0 + hi * 8);
    fa.q[0] = pa[0];
    fa.q[1] = pa[2];
    const v4u* pb = (const v4u*)(Bs + n * EPAD + e0 + hi * 8);
    fb.q[0] = pb[0];
    fb.q[1] = pb[2];
  };

  async_wait0();
  __syncthreads();

  // ---- main pipeline: gather(i+1) || GEMM(i) || scan(i-1) ----
  for (int lt = 0; lt < NLT; ++lt) {
    if (lt + 1 < NLT) gather_tile(lt + 1, Ab((lt + 1) & 1));

    if (wv < 7) {
      const unsigned short* Ad = Ab(lt & 1);
      float* TSs = TSb(lt & 1);
      const int n = wv * 16 + lw;
      v8f acc = {0.f, 0.f, 0.f, 0.f, 0.f, 0.f, 0.f, 0.f};
      FragBF fa[2], fb[2];
      load_frags(Ad, n, 0, fa[0], fb[0]);
#pragma unroll
      for (int k = 0; k < EPAD / 32; ++k) {
        const int cur = k & 1;
        if (k + 1 < EPAD / 32)
          load_frags(Ad, n, (k + 1) * 32, fa[cur ^ 1], fb[cur ^ 1]);
        acc = __builtin_amdgcn_wmma_f32_16x16x32_bf16(
            false, fa[cur].v, false, fb[cur].v, (short)0, acc, false, false);
      }
      const float bb = biasS[n];
#pragma unroll
      for (int r = 0; r < 8; ++r)
        TSs[(r + hi * 8) * DC + n] = acc[r] + bb;
    } else if (scanLane && lt > 0) {
      scan_tile(TSb((lt - 1) & 1));
    }

    async_wait0();
    __syncthreads();
  }

  // ---- epilogue: scan the last tile, emit scores ----
  if (scanLane) {
    scan_tile(TSb((NLT - 1) & 1));
    scores[b * Pp + pg * PG + ln] = sacc;
  }
}

// ---- K4: MLP + log_softmax -------------------------------------------------
__global__ void mlp_kernel(const float* __restrict__ scores,
                           const float* __restrict__ w1, const float* __restrict__ b1,
                           const float* __restrict__ w2, const float* __restrict__ b2,
                           const float* __restrict__ w3, const float* __restrict__ b3,
                           float* __restrict__ out) {
  __shared__ float s[Pp], h1[Hh], h2[Hh];
  const int b = blockIdx.x, t = threadIdx.x;
  for (int i = t; i < Pp; i += 128) s[i] = scores[b * Pp + i];
  __syncthreads();
  if (t < Hh) {
    float a = b1[t];
    for (int p = 0; p < Pp; ++p) a = fmaf(s[p], w1[p * Hh + t], a);
    h1[t] = fmaxf(a, 0.0f);
  }
  __syncthreads();
  if (t < Hh) {
    float a = b2[t];
    for (int j = 0; j < Hh; ++j) a = fmaf(h1[j], w2[j * Hh + t], a);
    h2[t] = fmaxf(a, 0.0f);
  }
  __syncthreads();
  if (t == 0) {
    float l0 = b3[0], l1 = b3[1];
    for (int j = 0; j < Hh; ++j) {
      l0 = fmaf(h2[j], w3[j * 2 + 0], l0);
      l1 = fmaf(h2[j], w3[j * 2 + 1], l1);
    }
    const float mx  = fmaxf(l0, l1);
    const float lse = mx + logf(expf(l0 - mx) + expf(l1 - mx));
    out[b * 2 + 0] = l0 - lse;
    out[b * 2 + 1] = l1 - lse;
  }
}

// ---------------------------------------------------------------------------
extern "C" void kernel_launch(void* const* d_in, const int* in_sizes, int n_in,
                              void* d_out, int out_size, void* d_ws, size_t ws_size,
                              hipStream_t stream) {
  const int*   docs  = (const int*)d_in[0];
  const float* emb   = (const float*)d_in[1];
  const float* diags = (const float*)d_in[2];
  const float* bias  = (const float*)d_in[3];
  const float* eps   = (const float*)d_in[4];
  const float* w1    = (const float*)d_in[5];
  const float* b1    = (const float*)d_in[6];
  const float* w2    = (const float*)d_in[7];
  const float* b2    = (const float*)d_in[8];
  const float* w3    = (const float*)d_in[9];
  const float* b3    = (const float*)d_in[10];
  float* out = (float*)d_out;

  char* ws = (char*)d_ws;
  unsigned short* embB  = (unsigned short*)(ws + WS_EMB);
  unsigned short* diagB = (unsigned short*)(ws + WS_DIAG);
  float*          score = (float*)(ws + WS_SCORE);

  cvt_emb_kernel<<<(V_ * EPAD + 255) / 256, 256, 0, stream>>>(emb, embB);
  cvt_diag_kernel<<<(Dd * EPAD + 255) / 256, 256, 0, stream>>>(diags, diagB);

  dim3 grid(Bb, NPG);
  sopa_kernel<<<grid, 256, LDS_TOTAL, stream>>>(docs, embB, diagB, bias, eps, score);

  mlp_kernel<<<Bb, 128, 0, stream>>>(score, w1, b1, w2, b2, w3, b3, out);
}
